// RoIHeads_90305982366406
// MI455X (gfx1250) — compile-verified, matrix operand order checked by reading, not run on hardware
//
#include <hip/hip_runtime.h>
#include <hip/hip_bf16.h>
#include <math.h>

// ---------------- problem constants (match reference) ----------------
#define BB 2
#define PP 1000
#define NCLS 91
#define CC 90            // NCLS - 1
#define DETS 100
#define SCORE_T 0.05f
#define NMS_T 0.5f
#define MINSZ 0.01f
#define IMGW 800.0f
#define IMGH 800.0f
#define BCLIP 4.135166556742356f   // log(1000/16)
#define NROWS (BB*PP)              // 2000

typedef __attribute__((ext_vector_type(2)))  float    v2f;
typedef __attribute__((ext_vector_type(8)))  float    v8f;
typedef __attribute__((ext_vector_type(16))) _Float16 v16h;

#if defined(__has_builtin)
#  if __has_builtin(__builtin_amdgcn_wmma_f32_16x16x4_f32)
#    define USE_WMMA_F32K4 1
#  endif
#endif

// =====================================================================
// Kernel 1: softmax (WMMA row-sums) + box decode + clip + validity.
// 8 waves/block, 16 proposal-rows per wave.  st layout [B][90][P],
// bt layout [B][90][P][4] (class-major, ready for per-(b,c) NMS).
// =====================================================================
__global__ __launch_bounds__(256)
void k1_decode_softmax(const float* __restrict__ logits,   // [2000][91]
                       const float* __restrict__ reg,      // [2000][364]
                       const float* __restrict__ props,    // [2000][4]
                       float* __restrict__ st,             // [B*90*1000]
                       float* __restrict__ bt)             // [B*90*1000*4]
{
    const int tid  = threadIdx.x;
    const int lane = tid & 31;
    const int wv   = tid >> 5;
    const int m    = lane & 15;     // row within 16-row tile
    const int hi   = lane >> 4;     // 0/1: which half-lane-group
    const int row0 = (blockIdx.x * 8 + wv) * 16;
    const int row  = row0 + m;
    const int rowc = (row < NROWS) ? row : (NROWS - 1);   // clamp: keep EXEC full

    __shared__ float s_sum[8][16];
    __shared__ float s_max[8][16];

    // ---- row max over 91 classes (2 lanes per row, wave32 shfl) ----
    const float* lrow = logits + (size_t)rowc * NCLS;
    float mx = -3.0e38f;
    for (int c0 = hi; c0 < NCLS; c0 += 2) mx = fmaxf(mx, lrow[c0]);
    mx = fmaxf(mx, __shfl_xor(mx, 16));
    if (hi == 0) s_max[wv][m] = mx;

    // ---- row-sum of exp(logit - max) via matrix core (B = ones) ----
    v8f acc = {};
#ifdef USE_WMMA_F32K4
    // A is 16x4 f32: lanes 0-15 hold K={kc,kc+1}, lanes 16-31 K={kc+2,kc+3}
    v2f bones; bones.x = 1.0f; bones.y = 1.0f;
    #pragma unroll
    for (int kc = 0; kc < 92; kc += 4) {
        const int k0 = kc + ((lane < 16) ? 0 : 2);
        v2f a;
        a.x = (k0     < NCLS) ? expf(lrow[k0]     - mx) : 0.0f;
        a.y = (k0 + 1 < NCLS) ? expf(lrow[k0 + 1] - mx) : 0.0f;
        acc = __builtin_amdgcn_wmma_f32_16x16x4_f32(
                  false, a, false, bones, (short)0, acc, false, false);
    }
#else
    // Fallback: f16 16x16x32 (codegen-confirmed builtin), 3 chunks of K=32.
    v16h bones;
    #pragma unroll
    for (int i = 0; i < 16; ++i) bones[i] = (_Float16)1.0f;
    #pragma unroll
    for (int kc = 0; kc < 96; kc += 32) {
        v16h a;
        #pragma unroll
        for (int v = 0; v < 8; ++v) {
            const int kb = kc + ((v < 4) ? 0 : 16) + ((lane >= 16) ? 8 : 0)
                         + 2 * (v & 3);
            const float e0 = (kb     < NCLS) ? expf(lrow[kb]     - mx) : 0.0f;
            const float e1 = (kb + 1 < NCLS) ? expf(lrow[kb + 1] - mx) : 0.0f;
            a[2 * v]     = (_Float16)e0;
            a[2 * v + 1] = (_Float16)e1;
        }
        acc = __builtin_amdgcn_wmma_f32_16x16x32_f16(
                  false, a, false, bones, (short)0, acc, false, false);
    }
#endif

    // ---- extract row-sums from C/D layout into LDS ----
    // C layout: VGPR r -> lanes 0-15: (M=r, N=lane); lanes 16-31: (M=8+r, N=lane-16)
    float rsv = 0.0f;
    #pragma unroll
    for (int r = 0; r < 8; ++r)
        if ((lane & 7) == r) rsv = acc[r];
    if (lane < 8)                      s_sum[wv][lane] = rsv;        // rows 0..7
    else if (lane >= 16 && lane < 24)  s_sum[wv][(lane - 16) + 8] = rsv; // rows 8..15
    __syncthreads();

    // ---- decode + clip + validity for 16 rows x 90 classes per wave ----
    for (int it = lane; it < 16 * CC; it += 32) {
        const int mm = it / CC;
        const int c1 = (it % CC) + 1;           // classes 1..90
        const int r  = row0 + mm;
        if (r >= NROWS) continue;
        const int b = r / PP, p = r % PP;

        const float rm   = s_max[wv][mm];
        const float rsum = s_sum[wv][mm];
        const float sc   = expf(logits[(size_t)r * NCLS + c1] - rm) / rsum;

        const float* pr = props + (size_t)r * 4;
        const float w  = pr[2] - pr[0];
        const float h  = pr[3] - pr[1];
        const float cx = pr[0] + 0.5f * w;
        const float cy = pr[1] + 0.5f * h;

        const float* cd = reg + (size_t)r * (NCLS * 4) + c1 * 4;
        const float dx = cd[0] * 10.0f;                 // / 0.1
        const float dy = cd[1] * 10.0f;
        const float dw = fminf(cd[2] * 5.0f, BCLIP);    // / 0.2, clipped
        const float dh = fminf(cd[3] * 5.0f, BCLIP);

        const float pcx = dx * w + cx;
        const float pcy = dy * h + cy;
        const float pw  = expf(dw) * w;
        const float ph  = expf(dh) * h;

        float x1 = fminf(fmaxf(pcx - 0.5f * pw, 0.0f), IMGW);
        float y1 = fminf(fmaxf(pcy - 0.5f * ph, 0.0f), IMGH);
        float x2 = fminf(fmaxf(pcx + 0.5f * pw, 0.0f), IMGW);
        float y2 = fminf(fmaxf(pcy + 0.5f * ph, 0.0f), IMGH);

        const float bw = x2 - x1, bh = y2 - y1;
        const bool valid = (sc > SCORE_T) && (bw >= MINSZ) && (bh >= MINSZ);

        const size_t o = ((size_t)b * CC + (c1 - 1)) * PP + p;
        st[o] = valid ? sc : -1.0f;
        float4 bx; bx.x = x1; bx.y = y1; bx.z = x2; bx.w = y2;
        *(float4*)(bt + o * 4) = bx;
    }
}

// =====================================================================
// Kernel 2: per-(b,c) bitonic sort (desc score, asc index ties = stable
// argsort) + greedy NMS, fully in LDS.  180 blocks x 256 threads.
// =====================================================================
__global__ __launch_bounds__(256)
void k2_sort_nms(float* __restrict__ st, float* __restrict__ bt)
{
    const int bc  = blockIdx.x;         // 0..179
    const int tid = threadIdx.x;

    __shared__ float         skey[1024];
    __shared__ int           sidx[1024];
    __shared__ float         sbox[PP * 4];
    __shared__ unsigned char keep[PP];

    float* stp = st + (size_t)bc * PP;
    float* btp = bt + (size_t)bc * PP * 4;

    for (int i = tid; i < 1024; i += 256) {
        skey[i] = (i < PP) ? stp[i] : -1.0f;
        sidx[i] = i;
    }
    __syncthreads();

    // bitonic sort: descending by key, ascending index on ties
    for (int k = 2; k <= 1024; k <<= 1) {
        for (int j = k >> 1; j > 0; j >>= 1) {
            for (int i = tid; i < 1024; i += 256) {
                const int ixj = i ^ j;
                if (ixj > i) {
                    const float ki = skey[i], kj = skey[ixj];
                    const int   vi = sidx[i], vj = sidx[ixj];
                    const bool before = (ki > kj) || (ki == kj && vi < vj);
                    const bool desc   = ((i & k) == 0);
                    if (desc ? !before : before) {
                        skey[i] = kj; skey[ixj] = ki;
                        sidx[i] = vj; sidx[ixj] = vi;
                    }
                }
            }
            __syncthreads();
        }
    }

    // gather boxes in sorted order; init keep = (score > 0)
    for (int s = tid; s < PP; s += 256) {
        const int o = sidx[s];
        const float4 bx = *(const float4*)(btp + (size_t)o * 4);
        sbox[s * 4 + 0] = bx.x; sbox[s * 4 + 1] = bx.y;
        sbox[s * 4 + 2] = bx.z; sbox[s * 4 + 3] = bx.w;
        keep[s] = (skey[s] > 0.0f) ? 1 : 0;
    }
    __syncthreads();

    // greedy NMS (sorted desc => once score <= 0, all remaining invalid)
    for (int i = 0; i < PP - 1; ++i) {
        if (skey[i] <= 0.0f) break;           // uniform (LDS broadcast)
        if (keep[i]) {
            const float ax1 = sbox[i * 4 + 0], ay1 = sbox[i * 4 + 1];
            const float ax2 = sbox[i * 4 + 2], ay2 = sbox[i * 4 + 3];
            const float aarea = (ax2 - ax1) * (ay2 - ay1);
            for (int jj = i + 1 + tid; jj < PP; jj += 256) {
                if (!keep[jj]) continue;
                const float bx1 = sbox[jj * 4 + 0], by1 = sbox[jj * 4 + 1];
                const float bx2 = sbox[jj * 4 + 2], by2 = sbox[jj * 4 + 3];
                const float iw = fmaxf(fminf(ax2, bx2) - fmaxf(ax1, bx1), 0.0f);
                const float ih = fmaxf(fminf(ay2, by2) - fmaxf(ay1, by1), 0.0f);
                const float inter = iw * ih;
                const float barea = (bx2 - bx1) * (by2 - by1);
                const float iou = inter / (aarea + barea - inter + 1e-9f);
                if (iou > NMS_T) keep[jj] = 0;
            }
        }
        __syncthreads();
    }
    __syncthreads();

    // write back sorted boxes + post-NMS scores (suppressed -> -1)
    for (int s = tid; s < PP; s += 256) {
        stp[s] = keep[s] ? skey[s] : -1.0f;
        float4 bx;
        bx.x = sbox[s * 4 + 0]; bx.y = sbox[s * 4 + 1];
        bx.z = sbox[s * 4 + 2]; bx.w = sbox[s * 4 + 3];
        *(float4*)(btp + (size_t)s * 4) = bx;
    }
}

// =====================================================================
// Kernel 3: per-batch top-100 (iterative argmax, ties -> lowest index,
// matching lax.top_k).  Mutates fin (workspace), regenerated each call.
// Outputs: boxes [B*100*4] | scores [B*100] | labels-as-float [B*100].
// =====================================================================
__global__ __launch_bounds__(256)
void k3_topk(float* __restrict__ fin,          // [B][90000]
             const float* __restrict__ sb,     // [B][90000][4]
             float* __restrict__ out)
{
    const int b   = blockIdx.x;
    const int tid = threadIdx.x;
    const int N   = CC * PP;                   // 90000

    __shared__ float bv[256];
    __shared__ int   bi[256];

    float*       f     = fin + (size_t)b * N;
    const float* boxes = sb  + (size_t)b * N * 4;
    float* ob = out;
    float* os = out + BB * DETS * 4;
    float* ol = out + BB * DETS * 4 + BB * DETS;

    for (int it = 0; it < DETS; ++it) {
        float mv = -3.0e38f;
        int   mi = 0;
        for (int idx = tid; idx < N; idx += 256) {
            const float v = f[idx];
            if (v > mv) { mv = v; mi = idx; }  // strided asc => first occurrence
        }
        bv[tid] = mv; bi[tid] = mi;
        __syncthreads();
        for (int s = 128; s > 0; s >>= 1) {
            if (tid < s) {
                const float v2 = bv[tid + s];
                const int   i2 = bi[tid + s];
                if (v2 > bv[tid] || (v2 == bv[tid] && i2 < bi[tid])) {
                    bv[tid] = v2; bi[tid] = i2;
                }
            }
            __syncthreads();
        }
        if (tid == 0) {
            const float v  = bv[0];
            const int   ix = bi[0];
            const bool det = (v > 0.0f);
            const int  o   = b * DETS + it;
            ob[o * 4 + 0] = det ? boxes[(size_t)ix * 4 + 0] : 0.0f;
            ob[o * 4 + 1] = det ? boxes[(size_t)ix * 4 + 1] : 0.0f;
            ob[o * 4 + 2] = det ? boxes[(size_t)ix * 4 + 2] : 0.0f;
            ob[o * 4 + 3] = det ? boxes[(size_t)ix * 4 + 3] : 0.0f;
            os[o] = det ? v : 0.0f;
            ol[o] = det ? (float)(ix / PP + 1) : 0.0f;
            f[ix] = -2.0f;                     // remove from candidate set
        }
        __syncthreads();
    }
}

// =====================================================================
extern "C" void kernel_launch(void* const* d_in, const int* in_sizes, int n_in,
                              void* d_out, int out_size, void* d_ws, size_t ws_size,
                              hipStream_t stream)
{
    (void)in_sizes; (void)n_in; (void)out_size;

    const float* logits = (const float*)d_in[0];   // [2000,91]
    const float* reg    = (const float*)d_in[1];   // [2000,364]
    const float* props  = (const float*)d_in[2];   // [2,1000,4]

    // workspace layout: st [180000] f32 | bt [180000*4] f32  (~3.6 MB)
    float* st = (float*)d_ws;
    float* bt = st + (size_t)BB * CC * PP;
    if (ws_size < (size_t)BB * CC * PP * 5 * sizeof(float)) return;

    // 16 blocks * 8 waves * 16 rows = 2048 >= 2000 rows
    k1_decode_softmax<<<16, 256, 0, stream>>>(logits, reg, props, st, bt);
    k2_sort_nms<<<BB * CC, 256, 0, stream>>>(st, bt);
    k3_topk<<<BB, 256, 0, stream>>>(st, bt, (float*)d_out);
}
